// LshModule_74466142978330
// MI455X (gfx1250) — 16-bit and 8-bit matrix operand order repaired and checked by reading, compiled, not run on hardware
//
#include <hip/hip_runtime.h>
#include <math.h>
#include <stdint.h>

typedef __attribute__((ext_vector_type(16))) _Float16 v16h;
typedef __attribute__((ext_vector_type(8)))  float    v8f;
typedef __attribute__((ext_vector_type(16))) int      v16i;

#define NUM_BITS 1024
#define DIMK     128
#define KTILES   (NUM_BITS / 128)
#define FRAG_BYTES 2048

#define FP8_ONE  ((unsigned char)0x38)

__global__ __launch_bounds__(256) void lsh_proj_bin(const float* __restrict__ emb,
                                                    const float* __restrict__ r,
                                                    unsigned char* __restrict__ codes,
                                                    int nRows)
{
    const int wave     = blockIdx.x * (blockDim.x >> 5) + (threadIdx.x >> 5);
    const int bitTiles = NUM_BITS / 16;
    const int rowTile  = wave / bitTiles;
    const int bitTile  = wave % bitTiles;
    if (rowTile * 16 >= nRows) return;

    const int lane = threadIdx.x & 31;
    const int n0   = lane & 15;
    const int hi   = lane >> 4;

    const float* arow = emb + (size_t)(rowTile * 16 + n0) * DIMK;
    const float* brow = r   + (size_t)(bitTile * 16 + n0) * DIMK;
    const int aoff = hi * 16;
    const int boff = hi * 16;

    v8f c = {};
#pragma unroll
    for (int kk = 0; kk < DIMK; kk += 32) {
        const float4* pa0 = reinterpret_cast<const float4*>(arow + kk + aoff);
        const float4* pa1 = reinterpret_cast<const float4*>(arow + kk + 8 + aoff);
        float4 f0 = pa0[0], f1 = pa0[1];
        float4 f2 = pa1[0], f3 = pa1[1];
        v16h a;
        a[0]  = (_Float16)f0.x; a[1]  = (_Float16)f0.y; a[2]  = (_Float16)f0.z; a[3]  = (_Float16)f0.w;
        a[4]  = (_Float16)f1.x; a[5]  = (_Float16)f1.y; a[6]  = (_Float16)f1.z; a[7]  = (_Float16)f1.w;
        a[8]  = (_Float16)f2.x; a[9]  = (_Float16)f2.y; a[10] = (_Float16)f2.z; a[11] = (_Float16)f2.w;
        a[12] = (_Float16)f3.x; a[13] = (_Float16)f3.y; a[14] = (_Float16)f3.z; a[15] = (_Float16)f3.w;

        const float4* pb = reinterpret_cast<const float4*>(brow + kk + boff);
        float4 g0 = pb[0], g1 = pb[1], g2 = pb[2], g3 = pb[3];
        v16h b;
        b[0]  = (_Float16)g0.x; b[1]  = (_Float16)g0.y; b[2]  = (_Float16)g0.z; b[3]  = (_Float16)g0.w;
        b[4]  = (_Float16)g1.x; b[5]  = (_Float16)g1.y; b[6]  = (_Float16)g1.z; b[7]  = (_Float16)g1.w;
        b[8]  = (_Float16)g2.x; b[9]  = (_Float16)g2.y; b[10] = (_Float16)g2.z; b[11] = (_Float16)g2.w;
        b[12] = (_Float16)g3.x; b[13] = (_Float16)g3.y; b[14] = (_Float16)g3.z; b[15] = (_Float16)g3.w;

        c = __builtin_amdgcn_wmma_f32_16x16x32_f16(false, a, false, b, (short)0, c, false, false);
    }

    unsigned char* base = codes + (size_t)(rowTile * 16 + hi * 8) * NUM_BITS
                                + (size_t)(bitTile * 16 + n0);
#pragma unroll
    for (int v = 0; v < 8; ++v)
        base[(size_t)v * NUM_BITS] = (c[v] > 0.0f) ? FP8_ONE : (unsigned char)0;
}

__global__ __launch_bounds__(256) void lsh_row_sums(const unsigned char* __restrict__ ucodes,
                                                    const unsigned char* __restrict__ vcodes,
                                                    int* __restrict__ su,
                                                    int* __restrict__ sv,
                                                    int n1, int n2)
{
    int tid = blockIdx.x * blockDim.x + threadIdx.x;
    if (tid >= n1 + n2) return;
    const uint4* p = (tid < n1)
        ? reinterpret_cast<const uint4*>(ucodes + (size_t)tid * NUM_BITS)
        : reinterpret_cast<const uint4*>(vcodes + (size_t)(tid - n1) * NUM_BITS);
    int s = 0;
#pragma unroll 8
    for (int i = 0; i < NUM_BITS / 16; ++i) {
        uint4 w = p[i];
        s += __builtin_popcount(w.x & 0x08080808u) + __builtin_popcount(w.y & 0x08080808u)
           + __builtin_popcount(w.z & 0x08080808u) + __builtin_popcount(w.w & 0x08080808u);
    }
    if (tid < n1) su[tid] = s; else sv[tid - n1] = s;
}

__global__ __launch_bounds__(256) void lsh_repack(const unsigned char* __restrict__ src,
                                                  unsigned char* __restrict__ dst,
                                                  int nRows, int isA)
{
    const int tid  = blockIdx.x * blockDim.x + threadIdx.x;
    const int j    = tid & 3;
    const int lane = (tid >> 2) & 31;
    const int frag = tid >> 7;
    const int rowTile = frag / KTILES;
    const int kTile   = frag % KTILES;
    if (rowTile * 16 >= nRows) return;

    const unsigned char* row = src + (size_t)(rowTile * 16 + (lane & 15)) * NUM_BITS
                                   + (size_t)kTile * 128;
    unsigned char* o = dst + (size_t)frag * FRAG_BYTES + (size_t)j * 512 + (size_t)lane * 16;

    if (isA) {
        const int c0 = (lane >> 4) * 16;
        uint64_t w0 = *reinterpret_cast<const uint64_t*>(row + 32 * j + c0);
        uint64_t w1 = *reinterpret_cast<const uint64_t*>(row + 32 * j + 8 + c0);
        reinterpret_cast<uint64_t*>(o)[0] = w0;
        reinterpret_cast<uint64_t*>(o)[1] = w1;
    } else {
        const int b0 = (lane >> 4) * 16;
        uint4 w = *reinterpret_cast<const uint4*>(row + 32 * j + b0);
        *reinterpret_cast<uint4*>(o) = w;
    }
}

__global__ __launch_bounds__(256) void lsh_hamm_cos(const unsigned char* __restrict__ uswz,
                                                    const unsigned char* __restrict__ vswz,
                                                    const int* __restrict__ su,
                                                    const int* __restrict__ sv,
                                                    const float* __restrict__ pi,
                                                    float* __restrict__ out,
                                                    int n1, int n2)
{
    const int wave   = blockIdx.x * (blockDim.x >> 5) + (threadIdx.x >> 5);
    const int nTiles = n2 / 32;
    const int mIdx   = wave / nTiles;
    const int nIdx   = wave % nTiles;
    const int rowBase = mIdx * 64;
    const int colBase = nIdx * 32;
    if (rowBase >= n1) return;

    const int lane = threadIdx.x & 31;
    const int n0   = lane & 15;
    const int hi   = lane >> 4;

    const unsigned char* au = uswz + (size_t)lane * 16;
    const unsigned char* bv = vswz + (size_t)lane * 16;

    v8f acc[8] = {};

    for (int kT = 0; kT < KTILES; ++kT) {
        v16i a[4];
#pragma unroll
        for (int m = 0; m < 4; ++m) {
            const unsigned char* f = au + (size_t)((mIdx * 4 + m) * KTILES + kT) * FRAG_BYTES;
            v16i t;
#pragma unroll
            for (int j = 0; j < 4; ++j) {
                int4 w = *reinterpret_cast<const int4*>(f + j * 512);
                t[4 * j] = w.x; t[4 * j + 1] = w.y; t[4 * j + 2] = w.z; t[4 * j + 3] = w.w;
            }
            a[m] = t;
        }
        v16i b[2];
#pragma unroll
        for (int n = 0; n < 2; ++n) {
            const unsigned char* f = bv + (size_t)((nIdx * 2 + n) * KTILES + kT) * FRAG_BYTES;
            v16i t;
#pragma unroll
            for (int j = 0; j < 4; ++j) {
                int4 w = *reinterpret_cast<const int4*>(f + j * 512);
                t[4 * j] = w.x; t[4 * j + 1] = w.y; t[4 * j + 2] = w.z; t[4 * j + 3] = w.w;
            }
            b[n] = t;
        }
#pragma unroll
        for (int m = 0; m < 4; ++m)
#pragma unroll
            for (int n = 0; n < 2; ++n)
                acc[m * 2 + n] = __builtin_amdgcn_wmma_f32_16x16x128_fp8_fp8(
                    a[m], b[n], (short)0, acc[m * 2 + n], false, false);
    }

    const float scale = pi[0] / (float)NUM_BITS;
#pragma unroll
    for (int m = 0; m < 4; ++m) {
        float srow[8];
#pragma unroll
        for (int v = 0; v < 8; ++v)
            srow[v] = (float)su[rowBase + m * 16 + hi * 8 + v];
#pragma unroll
        for (int n = 0; n < 2; ++n) {
            const int   col  = colBase + n * 16 + n0;
            const float scol = (float)sv[col];
            v8f c = acc[m * 2 + n];
#pragma unroll
            for (int v = 0; v < 8; ++v) {
                const float h   = srow[v] + scol - 2.0f * c[v];
                const int   row = rowBase + m * 16 + hi * 8 + v;
                out[(size_t)row * n2 + col] = cosf(scale * h);
            }
        }
    }
}

extern "C" void kernel_launch(void* const* d_in, const int* in_sizes, int n_in,
                              void* d_out, int out_size, void* d_ws, size_t ws_size,
                              hipStream_t stream) {
    const float* emb1 = (const float*)d_in[0];
    const float* emb2 = (const float*)d_in[1];
    const float* r    = (const float*)d_in[2];
    const float* pi   = (const float*)d_in[3];
    float* out = (float*)d_out;

    const int n1 = in_sizes[0] / DIMK;
    const int n2 = in_sizes[1] / DIMK;

    unsigned char* ws     = (unsigned char*)d_ws;
    unsigned char* ucodes = ws;
    unsigned char* vcodes = ucodes + (size_t)n1 * NUM_BITS;
    int* su = (int*)(vcodes + (size_t)n2 * NUM_BITS);
    int* sv = su + n1;
    unsigned char* uswz = (unsigned char*)(sv + n2);
    unsigned char* vswz = uswz + (size_t)n1 * NUM_BITS;

    {
        int tiles = (n1 / 16) * (NUM_BITS / 16);
        lsh_proj_bin<<<tiles / 8, 256, 0, stream>>>(emb1, r, ucodes, n1);
    }
    {
        int tiles = (n2 / 16) * (NUM_BITS / 16);
        lsh_proj_bin<<<tiles / 8, 256, 0, stream>>>(emb2, r, vcodes, n2);
    }
    {
        int total = n1 + n2;
        lsh_row_sums<<<(total + 255) / 256, 256, 0, stream>>>(ucodes, vcodes, su, sv, n1, n2);
    }
    {
        int threadsA = (n1 / 16) * KTILES * 128;
        lsh_repack<<<threadsA / 256, 256, 0, stream>>>(ucodes, uswz, n1, 1);
        int threadsB = (n2 / 16) * KTILES * 128;
        lsh_repack<<<threadsB / 256, 256, 0, stream>>>(vcodes, vswz, n2, 0);
    }
    {
        int waves = (n1 / 64) * (n2 / 32);
        lsh_hamm_cos<<<waves / 8, 256, 0, stream>>>(uswz, vswz, su, sv, pi, out, n1, n2);
    }
}
